// NeuralDecoder_52338471469443
// MI455X (gfx1250) — compile-verified, matrix operand order checked by reading, not run on hardware
//
#include <hip/hip_runtime.h>
#include <hip/hip_bf16.h>

// ---------------------------------------------------------------------------
// NeuralDecoder for MI455X (gfx1250, wave32).
// B=S=T=128, E=512, HID=1024, NH=16, DH=64, IN=1536, V=8192.
// All GEMMs: v_wmma_f32_16x16x32_f16, f16 A and W operands, f32 accumulate.
// Wave tile 32x64 (two M sub-tiles share each B fragment): 8 WMMA / 12 loads
// per k-step, no converts in the hot loop. Workspace ~0.67 GB.
// ---------------------------------------------------------------------------

typedef _Float16 v16h __attribute__((ext_vector_type(16)));
typedef float    v8f  __attribute__((ext_vector_type(8)));
typedef _Float16 h8   __attribute__((ext_vector_type(8)));

#define WMMA_F32_F16(a, b, c) \
  __builtin_amdgcn_wmma_f32_16x16x32_f16(false, (a), false, (b), (short)0, (c), false, false)

// ---------------------------------------------------------------------------
__global__ void cvt_f32_f16(const float* __restrict__ s, _Float16* __restrict__ d, size_t n) {
  size_t i = (size_t)blockIdx.x * 256 + threadIdx.x;
  if (i < n) d[i] = (_Float16)s[i];
}

// ---------------------------------------------------------------------------
// K-phase of the GEMM: A pointers for the two M sub-rows advance by 32 halves
// per step; W pointer for N-tile t is wbase + t*16*K.
// Fragment K ordering per ISA 16-bit A layout (lane<16: kb=0, lane>=16: kb=8;
// halves 0..7 = K kb..kb+7, halves 8..15 = K kb+16..kb+23). B mirrors A with
// the lane indexing the W row (= B column).
// ---------------------------------------------------------------------------
__device__ __forceinline__ void gemm_phase(const _Float16* __restrict__ a0p,
                                           const _Float16* __restrict__ a1p,
                                           const _Float16* __restrict__ wbase,
                                           size_t wstride16, int ksteps,
                                           v8f* acc0, v8f* acc1) {
  for (int i = 0; i < ksteps; ++i) {
    v16h a0, a1;
    *(h8*)&a0       = *(const h8*)a0p;
    *((h8*)&a0 + 1) = *(const h8*)(a0p + 16);
    *(h8*)&a1       = *(const h8*)a1p;
    *((h8*)&a1 + 1) = *(const h8*)(a1p + 16);
    const _Float16* wp = wbase;
#pragma unroll
    for (int t = 0; t < 4; ++t) {
      v16h b;
      *(h8*)&b       = *(const h8*)wp;
      *((h8*)&b + 1) = *(const h8*)(wp + 16);
      __builtin_prefetch((const void*)(wp + 32), 0, 1);   // next k-step of W
      acc0[t] = WMMA_F32_F16(a0, b, acc0[t]);
      acc1[t] = WMMA_F32_F16(a1, b, acc1[t]);
      wp += wstride16;
    }
    a0p += 32; a1p += 32; wbase += 32;
  }
}

// ---------------------------------------------------------------------------
// C[M,N] = A[M,K](f16) @ W[N,K](f16)^T + bias ; writes f32 (Cf) and/or f16 (Ch).
// A split at ksplit into A1 (stride ksplit, optionally [T,B]->[B,S] transposed)
// and A2 (stride K-ksplit). ksplit is a multiple of 32, so each 32-wide k-step
// lies entirely in one source. Wave tile 32x64; block 8 waves -> 64x256 tile;
// grid (N/256, M/64).
// ---------------------------------------------------------------------------
__global__ __launch_bounds__(256)
void gemm_wmma(const _Float16* __restrict__ A1, const _Float16* __restrict__ A2,
               int ksplit, int a1trans,
               const _Float16* __restrict__ W, const float* __restrict__ bias,
               float* __restrict__ Cf, _Float16* __restrict__ Ch,
               int M, int N, int K) {
  const int lane  = threadIdx.x & 31;
  const int wv    = threadIdx.x >> 5;
  const int mTile = blockIdx.y * 64 + (wv >> 2) * 32;
  const int nTile = blockIdx.x * 256 + (wv & 3) * 64;
  const int nl    = lane & 15;
  const int half  = lane >> 4;
  const int kb    = half * 8;

  v8f acc0[4], acc1[4];
#pragma unroll
  for (int t = 0; t < 4; ++t) {
    const float bv = bias ? bias[nTile + t * 16 + nl] : 0.0f;
    for (int r = 0; r < 8; ++r) { acc0[t][r] = bv; acc1[t][r] = bv; }
  }

  const int row0 = mTile + nl, row1 = row0 + 16;
  const int ar0 = a1trans ? ((row0 & 127) * 128 + (row0 >> 7)) : row0;
  const int ar1 = a1trans ? ((row1 & 127) * 128 + (row1 >> 7)) : row1;
  const size_t wstride16 = (size_t)16 * K;
  const _Float16* wbase = W + (size_t)(nTile + nl) * K + kb;

  if (ksplit >= K) {
    gemm_phase(A1 + (size_t)ar0 * ksplit + kb,
               A1 + (size_t)ar1 * ksplit + kb,
               wbase, wstride16, K >> 5, acc0, acc1);
  } else {
    gemm_phase(A1 + (size_t)ar0 * ksplit + kb,
               A1 + (size_t)ar1 * ksplit + kb,
               wbase, wstride16, ksplit >> 5, acc0, acc1);
    const int K2 = K - ksplit;
    gemm_phase(A2 + (size_t)row0 * K2 + kb,
               A2 + (size_t)row1 * K2 + kb,
               wbase + ksplit, wstride16, K2 >> 5, acc0, acc1);
  }

  // epilogue: branch hoisted, incremental row addressing (idx += N)
#pragma unroll
  for (int sub = 0; sub < 2; ++sub) {
    const v8f* acc = sub ? acc1 : acc0;
    const int mBase = mTile + sub * 16 + half * 8;
    if (Cf) {
#pragma unroll
      for (int t = 0; t < 4; ++t) {
        size_t idx = (size_t)mBase * N + nTile + t * 16 + nl;
#pragma unroll
        for (int r = 0; r < 8; ++r) { Cf[idx] = acc[t][r]; idx += N; }
      }
    }
    if (Ch) {
#pragma unroll
      for (int t = 0; t < 4; ++t) {
        size_t idx = (size_t)mBase * N + nTile + t * 16 + nl;
#pragma unroll
        for (int r = 0; r < 8; ++r) { Ch[idx] = (_Float16)acc[t][r]; idx += N; }
      }
    }
  }
}

// ---------------------------------------------------------------------------
// Attention core: one wave per (b,s). q/k/v are f16 (GEMM f16 outputs).
//   num[i,j] = q[b,i,:] . k[b,s,j,:]   (16x16x64 -> two WMMA k-steps)
//   scores  = w0*dot + w1*cos + w2*(-cdist); softmax over j,
//   out[i,d] = attn @ v[b,s]           (K=16 zero-padded to 32, 4 N-tiles)
// attn -> d_out (f32); out -> ctx_pre (f16) with the reference's scrambled
// reshape: ctx_pre[b, i*8 + s/16, (s%16)*64 + d].
// ---------------------------------------------------------------------------
__global__ __launch_bounds__(256)
void attn_core(const _Float16* __restrict__ qbuf, const _Float16* __restrict__ kbuf,
               const _Float16* __restrict__ vbuf, const float* __restrict__ score_w,
               float* __restrict__ attn_out, _Float16* __restrict__ ctx_pre) {
  __shared__ _Float16 atile[8][256];
  const int lane = threadIdx.x & 31;
  const int wv   = threadIdx.x >> 5;
  const int gid  = blockIdx.x * 8 + wv;
  const int b = gid >> 7, s = gid & 127;
  const int nl = lane & 15, half = lane >> 4, kb = half * 8;

  float w0 = score_w[0], w1 = score_w[1], w2 = score_w[2];
  float wm = fmaxf(w0, fmaxf(w1, w2));
  float e0 = __expf(w0 - wm), e1 = __expf(w1 - wm), e2 = __expf(w2 - wm);
  float es = e0 + e1 + e2;
  w0 = e0 / es; w1 = e1 / es; w2 = e2 / es;

  const _Float16* qrow = qbuf + (size_t)b * 1024;
  const _Float16* krow = kbuf + (size_t)(b * 128 + s) * 1024;
  const _Float16* vrow = vbuf + (size_t)(b * 128 + s) * 1024;

  // per-row sum of squares: lanes 0-15 -> q rows, lanes 16-31 -> k rows
  const _Float16* statp = (half == 0) ? (qrow + nl * 64) : (krow + nl * 64);
  float ss = 0.f;
  for (int i = 0; i < 64; i += 8) {
    h8 u = *(const h8*)(statp + i);
#pragma unroll
    for (int j = 0; j < 8; ++j) { const float f = (float)u[j]; ss += f * f; }
  }
  const float nr = sqrtf(ss);

  // num = q @ k^T  (two K=32 WMMA steps)
  v8f num;
  for (int r = 0; r < 8; ++r) num[r] = 0.f;
#pragma unroll
  for (int c = 0; c < 2; ++c) {
    v16h qa, ka;
    const _Float16* qp = qrow + nl * 64 + c * 32 + kb;
    const _Float16* kp = krow + nl * 64 + c * 32 + kb;
    *(h8*)&qa       = *(const h8*)qp;
    *((h8*)&qa + 1) = *(const h8*)(qp + 16);
    *(h8*)&ka       = *(const h8*)kp;
    *((h8*)&ka + 1) = *(const h8*)(kp + 16);
    num = WMMA_F32_F16(qa, ka, num);
  }

  // blend scores + softmax over j (lane groups of 16)
  const float skn = __shfl(ss, 16 + nl);
  const float knn = __shfl(nr, 16 + nl);
  float at[8];
#pragma unroll
  for (int r = 0; r < 8; ++r) {
    const int mi = r + half * 8;
    const float sqm = __shfl(ss, mi);
    const float qnm = __shfl(nr, mi);
    const float nm = num[r];
    const float dotv = nm * 0.125f;                       // 1/sqrt(64)
    const float cosv = nm / fmaxf(qnm * knn, 1e-8f);
    const float d2 = sqm + skn - 2.f * nm;
    const float dist = -sqrtf(fmaxf(d2, 0.f));
    at[r] = w0 * dotv + w1 * cosv + w2 * dist;
  }
#pragma unroll
  for (int r = 0; r < 8; ++r) {
    float mx = at[r];
    for (int mask = 8; mask >= 1; mask >>= 1) mx = fmaxf(mx, __shfl_xor(mx, mask));
    float e = __expf(at[r] - mx);
    float sm = e;
    for (int mask = 8; mask >= 1; mask >>= 1) sm += __shfl_xor(sm, mask);
    const float a = e / sm;
    const int mi = r + half * 8;
    attn_out[((size_t)(b * 128 + s) * 16 + mi) * 16 + nl] = a;
    atile[wv][mi * 16 + nl] = (_Float16)a;
  }
  __syncthreads();

  // A-frag of attn (K = j, 16 real + 16 zero pad)
  v16h aa;
#pragma unroll
  for (int jj = 0; jj < 8; ++jj) aa[jj] = atile[wv][nl * 16 + kb + jj];
#pragma unroll
  for (int jj = 8; jj < 16; ++jj) aa[jj] = (_Float16)0.f;

  // out = attn @ v : 4 N-tiles of 16 columns of DH
#pragma unroll
  for (int t = 0; t < 4; ++t) {
    v16h vb;
#pragma unroll
    for (int jj = 0; jj < 8; ++jj) vb[jj] = vrow[(kb + jj) * 64 + t * 16 + nl];
#pragma unroll
    for (int jj = 8; jj < 16; ++jj) vb[jj] = (_Float16)0.f;
    v8f o;
    for (int r = 0; r < 8; ++r) o[r] = 0.f;
    o = WMMA_F32_F16(aa, vb, o);
    const int ccol = (s & 15) * 64 + t * 16 + nl;
#pragma unroll
    for (int r = 0; r < 8; ++r) {
      const int mi = r + half * 8;                        // head index i
      ctx_pre[(size_t)b * 131072 + (size_t)(mi * 8 + (s >> 4)) * 1024 + ccol] =
          (_Float16)o[r];
    }
  }
}

// ---------------------------------------------------------------------------
// xs[t,b,0:512]    = emb[x[b,t]] + pe[b]   (B==S broadcast quirk: pe row = b!)
// xs[t,b,512:1536] = ctx[b,t,:]            (xs stored f16 for the gi0 GEMM)
// ---------------------------------------------------------------------------
__global__ void embed_concat(const int* __restrict__ x, const float* __restrict__ emb,
                             const float* __restrict__ pe, const float* __restrict__ ctx,
                             _Float16* __restrict__ xs) {
  size_t idx = (size_t)blockIdx.x * 256 + threadIdx.x;   // < 128*128*1536
  const int c = (int)(idx % 1536);
  const size_t tb = idx / 1536;
  const int b = (int)(tb & 127);
  const int t = (int)(tb >> 7);
  float v;
  if (c < 512) {
    const int tok = x[b * 128 + t];
    v = emb[(size_t)tok * 512 + c] + pe[(size_t)b * 512 + c];
  } else {
    v = ctx[(size_t)b * 131072 + (size_t)t * 1024 + (c - 512)];
  }
  xs[idx] = (_Float16)v;
}

// ---------------------------------------------------------------------------
// GRU gate fuse: gi (f16) + gh (f32) + h (f32) -> h (f32), h16 (f16), ys (f16)
// ---------------------------------------------------------------------------
__global__ void gru_gate(const _Float16* __restrict__ gi, const float* __restrict__ gh,
                         float* __restrict__ h, _Float16* __restrict__ h16,
                         _Float16* __restrict__ ys) {
  const int idx = blockIdx.x * 256 + threadIdx.x;        // < 128*1024
  const int b = idx >> 10, c = idx & 1023;
  const size_t g = (size_t)b * 3072 + c;
  const float ir = (float)gi[g], iz = (float)gi[g + 1024], inn = (float)gi[g + 2048];
  const float hr = gh[g], hz = gh[g + 1024], hn = gh[g + 2048];
  const float r = 1.f / (1.f + __expf(-(ir + hr)));
  const float z = 1.f / (1.f + __expf(-(iz + hz)));
  const float n = tanhf(inn + r * hn);
  const float out = (1.f - z) * n + z * h[idx];
  h[idx] = out;
  h16[idx] = (_Float16)out;
  ys[idx] = (_Float16)out;
}

// ---------------------------------------------------------------------------
// LayerNorm + ELU; one wave per 1024-wide row; f32 in, f16 out (W2's A).
// ---------------------------------------------------------------------------
__global__ __launch_bounds__(256)
void ln_elu(const float* __restrict__ h, const float* __restrict__ g,
            const float* __restrict__ bt, _Float16* __restrict__ out) {
  const int lane = threadIdx.x & 31, wv = threadIdx.x >> 5;
  const size_t row = (size_t)blockIdx.x * 8 + wv;
  const float* p = h + row * 1024;
  float x[32];
  float s = 0.f;
  for (int i = 0; i < 32; i += 4) {
    float4 u = *(const float4*)(p + lane * 32 + i);
    x[i] = u.x; x[i + 1] = u.y; x[i + 2] = u.z; x[i + 3] = u.w;
    s += u.x + u.y + u.z + u.w;
  }
  for (int mask = 16; mask >= 1; mask >>= 1) s += __shfl_xor(s, mask);
  const float mu = s * (1.f / 1024.f);
  float v = 0.f;
  for (int i = 0; i < 32; ++i) { const float d = x[i] - mu; v += d * d; }
  for (int mask = 16; mask >= 1; mask >>= 1) v += __shfl_xor(v, mask);
  const float inv = rsqrtf(v * (1.f / 1024.f) + 1e-5f);
  for (int i = 0; i < 32; ++i) {
    const int c = lane * 32 + i;
    float y = (x[i] - mu) * inv * g[c] + bt[c];
    y = y > 0.f ? y : (__expf(y) - 1.f);
    out[row * 1024 + c] = (_Float16)y;
  }
}

// ---------------------------------------------------------------------------
extern "C" void kernel_launch(void* const* d_in, const int* in_sizes, int n_in,
                              void* d_out, int out_size, void* d_ws, size_t ws_size,
                              hipStream_t stream) {
  (void)in_sizes; (void)n_in; (void)out_size; (void)ws_size;
  const int*   x      = (const int*)d_in[0];
  const float* memory = (const float*)d_in[1];
  const float* hidden = (const float*)d_in[2];
  const float* pe     = (const float*)d_in[3];
  const float* emb    = (const float*)d_in[4];
  const float* Wq = (const float*)d_in[5];  const float* bq = (const float*)d_in[6];
  const float* Wk = (const float*)d_in[7];  const float* bk = (const float*)d_in[8];
  const float* Wv = (const float*)d_in[9];  const float* bv = (const float*)d_in[10];
  const float* Wo = (const float*)d_in[11]; const float* bo = (const float*)d_in[12];
  const float* score_w = (const float*)d_in[13];
  const float* Wih0 = (const float*)d_in[14]; const float* Whh0 = (const float*)d_in[15];
  const float* bih0 = (const float*)d_in[16]; const float* bhh0 = (const float*)d_in[17];
  const float* Wih1 = (const float*)d_in[18]; const float* Whh1 = (const float*)d_in[19];
  const float* bih1 = (const float*)d_in[20]; const float* bhh1 = (const float*)d_in[21];
  const float* W1 = (const float*)d_in[22];  const float* b1 = (const float*)d_in[23];
  const float* ln_g = (const float*)d_in[24]; const float* ln_b = (const float*)d_in[25];
  const float* W2 = (const float*)d_in[26];  const float* b2 = (const float*)d_in[27];

  float* outF     = (float*)d_out;
  float* logits   = outF;                       // [128,128,8192]
  float* attn_out = outF + 134217728ull;        // [128,128,16,16]
  float* hid_out  = outF + 138412032ull;        // [2,128,1024]

  // ---- workspace carve (~0.67 GB) ----
  char* wsb = (char*)d_ws; size_t off = 0;
  auto take = [&](size_t bytes) -> char* {
    char* p = wsb + off; off += (bytes + 255) & ~(size_t)255; return p;
  };
  _Float16* Wq16  = (_Float16*)take(1048576ull * 2);
  _Float16* Wk16  = (_Float16*)take(1048576ull * 2);
  _Float16* Wv16  = (_Float16*)take(1048576ull * 2);
  _Float16* Wo16  = (_Float16*)take(1048576ull * 2);
  _Float16* Wih0h = (_Float16*)take(4718592ull * 2);
  _Float16* Whh0h = (_Float16*)take(3145728ull * 2);
  _Float16* Wih1h = (_Float16*)take(3145728ull * 2);
  _Float16* Whh1h = (_Float16*)take(3145728ull * 2);
  _Float16* W1h   = (_Float16*)take(2097152ull * 2);
  _Float16* W2h   = (_Float16*)take(8388608ull * 2);
  _Float16* memh    = (_Float16*)take(16777216ull * 2);   // memory f16
  _Float16* qbuf    = (_Float16*)take(131072ull * 2);
  _Float16* kbuf    = (_Float16*)take(16777216ull * 2);
  _Float16* vbuf    = (_Float16*)take(16777216ull * 2);
  _Float16* ctx_pre = (_Float16*)take(16777216ull * 2);
  float*    ctx     = (float*)take(16777216ull * 4);      // f32 (embed reads it)
  _Float16* ctxh    = (_Float16*)take(16777216ull * 2);   // f16 (W1's A2)
  _Float16* xs      = (_Float16*)take(25165824ull * 2);
  _Float16* gi0     = (_Float16*)take(50331648ull * 2);
  _Float16* gi1     = (_Float16*)take(50331648ull * 2);
  _Float16* ys0h    = (_Float16*)take(16777216ull * 2);
  _Float16* ys1h    = (_Float16*)take(16777216ull * 2);
  float*    h0buf   = (float*)take(131072ull * 4);
  float*    h1buf   = (float*)take(131072ull * 4);
  _Float16* h0h     = (_Float16*)take(131072ull * 2);
  _Float16* h1h     = (_Float16*)take(131072ull * 2);
  float*    ghbuf   = (float*)take(393216ull * 4);
  float*    hbuf    = (float*)take(16777216ull * 4);
  _Float16* h16buf  = (_Float16*)take(16777216ull * 2);

  auto cvt = [&](const float* s, _Float16* d, size_t n) {
    cvt_f32_f16<<<(unsigned)((n + 255) / 256), 256, 0, stream>>>(s, d, n);
  };
  auto gemm = [&](const _Float16* A1, const _Float16* A2, int ksplit, int trans,
                  const _Float16* W, const float* bias, float* Cf, _Float16* Ch,
                  int M, int N, int K) {
    dim3 grid(N / 256, M / 64);
    gemm_wmma<<<grid, 256, 0, stream>>>(A1, A2, ksplit, trans, W, bias, Cf, Ch, M, N, K);
  };

  // ---- f16 conversions (weights L2-resident afterwards) ----
  cvt(Wq, Wq16, 1048576);   cvt(Wk, Wk16, 1048576);
  cvt(Wv, Wv16, 1048576);   cvt(Wo, Wo16, 1048576);
  cvt(Wih0, Wih0h, 4718592); cvt(Whh0, Whh0h, 3145728);
  cvt(Wih1, Wih1h, 3145728); cvt(Whh1, Whh1h, 3145728);
  cvt(W1, W1h, 2097152);    cvt(W2, W2h, 8388608);
  cvt(memory, memh, 16777216);
  cvt(hidden, h0h, 131072);
  cvt(hidden + 131072, h1h, 131072);

  // ---- attention ----
  gemm(memh, memh, 1024, 0, Wk16, bk, nullptr, kbuf, 16384, 1024, 1024);
  gemm(memh, memh, 1024, 0, Wv16, bv, nullptr, vbuf, 16384, 1024, 1024);
  gemm(h1h, h1h, 1024, 0, Wq16, bq, nullptr, qbuf, 128, 1024, 1024);
  attn_core<<<2048, 256, 0, stream>>>(qbuf, kbuf, vbuf, score_w, attn_out, ctx_pre);
  gemm(ctx_pre, ctx_pre, 1024, 0, Wo16, bo, ctx, ctxh, 16384, 1024, 1024);

  // ---- embed + concat -> xs [T,B,1536] (f16) ----
  embed_concat<<<98304, 256, 0, stream>>>(x, emb, pe, ctx, xs);

  // ---- GRU layer 0: batched gi, sequential gh ----
  gemm(xs, xs, 1536, 0, Wih0h, bih0, nullptr, gi0, 16384, 3072, 1536);
  hipMemcpyAsync(h0buf, hidden, 131072ull * 4, hipMemcpyDeviceToDevice, stream);
  hipMemcpyAsync(h1buf, hidden + 131072, 131072ull * 4, hipMemcpyDeviceToDevice, stream);
  for (int t = 0; t < 128; ++t) {
    gemm(h0h, h0h, 1024, 0, Whh0h, bhh0, ghbuf, nullptr, 128, 3072, 1024);
    gru_gate<<<512, 256, 0, stream>>>(gi0 + (size_t)t * 393216, ghbuf, h0buf, h0h,
                                      ys0h + (size_t)t * 131072);
  }
  // ---- GRU layer 1 ----
  gemm(ys0h, ys0h, 1024, 0, Wih1h, bih1, nullptr, gi1, 16384, 3072, 1024);
  for (int t = 0; t < 128; ++t) {
    gemm(h1h, h1h, 1024, 0, Whh1h, bhh1, ghbuf, nullptr, 128, 3072, 1024);
    gru_gate<<<512, 256, 0, stream>>>(gi1 + (size_t)t * 393216, ghbuf, h1buf, h1h,
                                      ys1h + (size_t)t * 131072);
  }

  // ---- h = concat(rnn_out[b,s,:], ctx[b,s,:]) @ W1^T + b1 (split-A, ys1 transposed) ----
  gemm(ys1h, ctxh, 1024, 1, W1h, b1, hbuf, nullptr, 16384, 1024, 2048);

  // ---- LayerNorm + ELU -> f16 ----
  ln_elu<<<2048, 256, 0, stream>>>(hbuf, ln_g, ln_b, h16buf);

  // ---- logits = elu(h) @ W2^T + b2 ----
  gemm(h16buf, h16buf, 1024, 0, W2h, b2, logits, nullptr, 16384, 8192, 1024);

  // ---- new_hidden ----
  hipMemcpyAsync(hid_out, h0buf, 131072ull * 4, hipMemcpyDeviceToDevice, stream);
  hipMemcpyAsync(hid_out + 131072, h1buf, 131072ull * 4, hipMemcpyDeviceToDevice, stream);
}
